// GlobalLocalModel_2576980377908
// MI455X (gfx1250) — matrix operand order repaired and checked by reading, not compiled, not run on hardware
//
#include <hip/hip_runtime.h>
#include <hip/hip_bf16.h>

typedef __attribute__((ext_vector_type(16))) _Float16 v16h;
typedef __attribute__((ext_vector_type(8)))  _Float16 v8h;
typedef __attribute__((ext_vector_type(8)))  float    v8f;

#define B_    16
#define KSEL  16
#define PS_   16
#define D_    256
#define S_    256
#define NH_   4
#define DH_   64
#define DFF_  1024
#define HGT   1024
#define WID   1024
#define NL_   2

// ---- d_out offsets (float elements), in reference return order ----
#define OUT_COARSE 0u          // 16*1*256*256 = 1048576
#define OUT_PATCH  1048576u    // 16*16*1*16*16 = 65536
#define OUT_PLAB   1114112u    // 65536
#define OUT_COORD  1179648u    // 16*16*2 = 512
#define OUT_PLOG   1180160u    // 65536
#define OUT_FINAL  1245696u    // 16*1*1024*1024 = 16777216

// ---- workspace offsets (bytes), all 256B aligned ----
#define WS_SCORES   0u           // 16*4096*4        = 262144
#define WS_IDX      262144u      // 256*4            = 1024
#define WS_LOGITS   263168u      // 4096*4           = 16384
#define WS_TOKF     279552u      // 4096*256*4       = 4194304
#define WS_TOKH     4473856u     // 4096*256*2       = 2097152
#define WS_ATTNF    6571008u     // 4194304
#define WS_ATTNH    10765312u    // 2097152
#define WS_SCRF     12862464u    // 4194304
#define WS_SCRH     17056768u    // 2097152
#define WS_BIGF     19153920u    // 16777216 (qkv f32 12MB / ff-hidden f32 16MB, phases disjoint)
#define WS_BIGH     35931136u    // 8388608  (ff-hidden f16)
#define WS_W_QKV    44319744u    // 2*768*256*2  = 786432
#define WS_W_OUT    45106176u    // 2*256*256*2  = 262144
#define WS_W_FF1    45368320u    // 2*1024*256*2 = 1048576
#define WS_W_FF2    46416896u    // 2*256*1024*2 = 1048576
#define WS_W_H1     47465472u    // 256*256*2    = 131072   (end 47596544)

// ===================== coarse = avg_pool4(labels) =====================
__global__ __launch_bounds__(256) void coarse_kernel(const float* __restrict__ labels,
                                                     float* __restrict__ coarse) {
  int tid = blockIdx.x * 256 + threadIdx.x;        // B*256*256
  int b = tid >> 16, cy = (tid >> 8) & 255, cx = tid & 255;
  const float* L = labels + ((size_t)b * HGT + cy * 4) * WID + cx * 4;
  float s = 0.f;
#pragma unroll
  for (int i = 0; i < 4; ++i)
#pragma unroll
    for (int j = 0; j < 4; ++j) s += L[i * WID + j];
  coarse[tid] = s * 0.0625f;
}

// ===================== per-patch scores from coarse =====================
__global__ __launch_bounds__(256) void scores_kernel(const float* __restrict__ coarse,
                                                     float* __restrict__ scores) {
  int tid = blockIdx.x * 256 + threadIdx.x;        // B*4096
  int b = tid >> 12, p = tid & 4095;
  int ph = p >> 6, pw = p & 63;
  const float* C = coarse + (size_t)b * 65536 + (ph * 4) * 256 + pw * 4;
  float s = 0.f;
#pragma unroll
  for (int i = 0; i < 4; ++i)
#pragma unroll
    for (int j = 0; j < 4; ++j) s += C[i * 256 + j];
  scores[tid] = s * 0.0625f;
}

// ===================== iterative top-16 per batch =====================
__global__ __launch_bounds__(256) void topk_kernel(float* __restrict__ scores,
                                                   int* __restrict__ idx,
                                                   float* __restrict__ coords) {
  int b = blockIdx.x, t = threadIdx.x;
  float* S = scores + b * 4096;
  __shared__ float bv[256];
  __shared__ int   bi[256];
  for (int sel = 0; sel < KSEL; ++sel) {
    float best = -3.4e38f; int besti = 1 << 30;
    for (int i = t; i < 4096; i += 256) {
      float v = S[i];
      if (v > best || (v == best && i < besti)) { best = v; besti = i; }
    }
    bv[t] = best; bi[t] = besti; __syncthreads();
    for (int o = 128; o; o >>= 1) {
      if (t < o) {
        if (bv[t + o] > bv[t] || (bv[t + o] == bv[t] && bi[t + o] < bi[t])) {
          bv[t] = bv[t + o]; bi[t] = bi[t + o];
        }
      }
      __syncthreads();
    }
    if (t == 0) {
      int w = bi[0];
      idx[b * KSEL + sel] = w;
      S[w] = -3.4e38f;
      coords[(b * KSEL + sel) * 2 + 0] = (float)((w >> 6) * PS_);
      coords[(b * KSEL + sel) * 2 + 1] = (float)((w & 63) * PS_);
    }
    __syncthreads();
  }
}

// ===================== gather selected patches =====================
__global__ __launch_bounds__(256) void gather_kernel(const float* __restrict__ image,
                                                     const float* __restrict__ labels,
                                                     const int* __restrict__ idx,
                                                     float* __restrict__ patches,
                                                     float* __restrict__ plab) {
  int bk = blockIdx.x, b = bk >> 4;
  int p = idx[bk], ph = p >> 6, pw = p & 63;
  int t = threadIdx.x, y = t >> 4, x = t & 15;
  size_t src = ((size_t)b * HGT + ph * PS_ + y) * WID + pw * PS_ + x;
  patches[(size_t)bk * 256 + t] = image[src];
  plab[(size_t)bk * 256 + t]    = labels[src];
}

// ===================== f32 -> f16 cast =====================
__global__ __launch_bounds__(256) void cast16_kernel(const float* __restrict__ s,
                                                     _Float16* __restrict__ d, int n) {
  int i = blockIdx.x * 256 + threadIdx.x;
  if (i < n) d[i] = (_Float16)s[i];
}

// ===================== patch embed + pos embed =====================
__global__ __launch_bounds__(256) void embed_kernel(const float* __restrict__ patches,
                                                    const float* __restrict__ pe_w,
                                                    const float* __restrict__ pe_b,
                                                    const float* __restrict__ pos,
                                                    float* __restrict__ tokf,
                                                    _Float16* __restrict__ tokh) {
  int tok = blockIdx.x;                            // 4096 tokens: bk*16 + (i*4+j)
  int bk = tok >> 4, ij = tok & 15;
  int i = ij >> 2, j = ij & 3;
  __shared__ float px[16];
  int t = threadIdx.x;
  if (t < 16) {
    int a = t >> 2, bb = t & 3;
    px[t] = patches[(size_t)bk * 256 + (i * 4 + a) * 16 + (j * 4 + bb)];
  }
  __syncthreads();
  float acc = pe_b[t];
#pragma unroll
  for (int e = 0; e < 16; ++e) acc = fmaf(pe_w[t * 16 + e], px[e], acc);
  acc += pos[ij * D_ + t];
  tokf[(size_t)tok * D_ + t] = acc;
  tokh[(size_t)tok * D_ + t] = (_Float16)acc;
}

// ===================== WMMA GEMM: Y[M,N] = act(X[M,K] * W[N,K]^T + bias) ==============
// One wave computes a 32x64 output block (MB=2 x NB=4 tiles of 16x16).
// Per K-step: 4 A-loads + 8 B-loads (all issued up front, b128 each) feed 8
// v_wmma_f32_16x16x32_f16 ops, so the waitcnt inserter can overlap loads with
// the WMMA chain. Requires M % 32 == 0, N % 64 == 0, K % 32 == 0 (true for all
// GEMMs here); EXEC stays all-ones as WMMA requires.
__global__ __launch_bounds__(32) void gemm_wmma_kernel(const _Float16* __restrict__ X,
                                                       const _Float16* __restrict__ W,
                                                       const float* __restrict__ bias,
                                                       float* __restrict__ Y,
                                                       int M, int N, int Kd, int act) {
  int ntn = N >> 6;                 // strips of 64 columns
  int tile = blockIdx.x;
  int tm = (tile / ntn) << 5;       // 32 rows
  int tn = (tile % ntn) << 6;       // 64 cols
  int lane = threadIdx.x & 31;
  int hl  = lane >> 4;              // lane-half selector
  int l16 = lane & 15;
  // A: row = tm + (mb*16) + l16, K packing: hl*8 + {0..7}, +16..+23
  const _Float16* xrow0 = X + (size_t)(tm + l16) * Kd + (hl << 3);
  const _Float16* xrow1 = xrow0 + (size_t)16 * Kd;
  // B: col = tn + nb*16 + l16 (== W row), K packing: hl*8 + {0..7}, +16..+23
  const _Float16* wrow = W + (size_t)(tn + l16) * Kd + (hl << 3);
  size_t wstep = (size_t)16 * Kd;   // advance 16 columns (W rows)
  v8f acc[2][4] = {{{}, {}, {}, {}}, {{}, {}, {}, {}}};
  for (int k0 = 0; k0 < Kd; k0 += 32) {
    // ---- issue all loads for this K-step first ----
    v8h a0lo = *(const v8h*)(xrow0 + k0);
    v8h a0hi = *(const v8h*)(xrow0 + k0 + 16);
    v8h a1lo = *(const v8h*)(xrow1 + k0);
    v8h a1hi = *(const v8h*)(xrow1 + k0 + 16);
    v8h b0[4], b1[4];
#pragma unroll
    for (int nb = 0; nb < 4; ++nb) {
      const _Float16* wr = wrow + (size_t)nb * wstep;
      b0[nb] = *(const v8h*)(wr + k0);
      b1[nb] = *(const v8h*)(wr + k0 + 16);
    }
    v16h A0, A1;
#pragma unroll
    for (int i = 0; i < 8; ++i) {
      A0[i] = a0lo[i]; A0[i + 8] = a0hi[i];
      A1[i] = a1lo[i]; A1[i + 8] = a1hi[i];
    }
    // ---- 8 WMMAs consuming the in-flight loads ----
#pragma unroll
    for (int nb = 0; nb < 4; ++nb) {
      v16h Bm;
#pragma unroll
      for (int i = 0; i < 8; ++i) { Bm[i] = b0[nb][i]; Bm[i + 8] = b1[nb][i]; }
      acc[0][nb] = __builtin_amdgcn_wmma_f32_16x16x32_f16(false, A0, false, Bm,
                                                          (short)0, acc[0][nb], false, false);
      acc[1][nb] = __builtin_amdgcn_wmma_f32_16x16x32_f16(false, A1, false, Bm,
                                                          (short)0, acc[1][nb], false, false);
    }
  }
#pragma unroll
  for (int mb = 0; mb < 2; ++mb) {
#pragma unroll
    for (int nb = 0; nb < 4; ++nb) {
      int col = tn + nb * 16 + l16;
      float bv = bias ? bias[col] : 0.f;
#pragma unroll
      for (int v = 0; v < 8; ++v) {
        int row = tm + mb * 16 + v + (hl << 3);
        float val = acc[mb][nb][v] + bv;
        if (act == 1) val = 0.5f * val * (1.f + erff(val * 0.70710678118654752f));
        Y[(size_t)row * N + col] = val;
      }
    }
  }
}

// ===================== attention (one block per b,h,q) =====================
__global__ __launch_bounds__(256) void attn_kernel(const float* __restrict__ qkv,
                                                   float* __restrict__ attn) {
  int blk = blockIdx.x;                // b*1024 + h*256 + q
  int q = blk & (S_ - 1);
  int h = (blk >> 8) & (NH_ - 1);
  int b = blk >> 10;
  const float* base = qkv + (size_t)b * S_ * (3 * D_);
  const float* Q = base + (size_t)q * (3 * D_) + h * DH_;
  __shared__ float sc[S_];
  __shared__ float red[256];
  __shared__ float part[4][DH_];
  int t = threadIdx.x;                 // key position
  const float* Krow = base + (size_t)t * (3 * D_) + D_ + h * DH_;
  float s = 0.f;
#pragma unroll 8
  for (int d = 0; d < DH_; ++d) s += Q[d] * Krow[d];
  s *= 0.125f;                         // 1/sqrt(64)
  sc[t] = s; red[t] = s; __syncthreads();
  for (int o = 128; o; o >>= 1) { if (t < o) red[t] = fmaxf(red[t], red[t + o]); __syncthreads(); }
  float mx = red[0]; __syncthreads();
  float e = expf(s - mx);
  sc[t] = e; red[t] = e; __syncthreads();
  for (int o = 128; o; o >>= 1) { if (t < o) red[t] += red[t + o]; __syncthreads(); }
  float inv = 1.f / red[0];
  int d = t & (DH_ - 1), ch = t >> 6;
  const float* V = base + 2 * D_ + h * DH_ + d;
  float o = 0.f;
  int k0 = ch * 64;
  for (int k = k0; k < k0 + 64; ++k) o += sc[k] * V[(size_t)k * (3 * D_)];
  part[ch][d] = o; __syncthreads();
  if (t < DH_) {
    float r = (part[0][t] + part[1][t] + part[2][t] + part[3][t]) * inv;
    attn[((size_t)b * S_ + q) * D_ + h * DH_ + t] = r;
  }
}

// ===================== residual add (optional) + LayerNorm =====================
__global__ __launch_bounds__(256) void addln_kernel(const float* __restrict__ x,
                                                    const float* __restrict__ resid,
                                                    const float* __restrict__ gamma,
                                                    const float* __restrict__ beta,
                                                    float* __restrict__ outf,
                                                    _Float16* __restrict__ outh) {
  int row = blockIdx.x, t = threadIdx.x;
  float v = x[(size_t)row * D_ + t];
  if (resid) v += resid[(size_t)row * D_ + t];
  __shared__ float red[256];
  red[t] = v; __syncthreads();
  for (int o = 128; o; o >>= 1) { if (t < o) red[t] += red[t + o]; __syncthreads(); }
  float mean = red[0] * (1.f / D_); __syncthreads();
  float c = v - mean;
  red[t] = c * c; __syncthreads();
  for (int o = 128; o; o >>= 1) { if (t < o) red[t] += red[t + o]; __syncthreads(); }
  float var = red[0] * (1.f / D_);
  float r = c * rsqrtf(var + 1e-5f) * gamma[t] + beta[t];
  outf[(size_t)row * D_ + t] = r;
  outh[(size_t)row * D_ + t] = (_Float16)r;
}

// ===================== head D->1 projection =====================
__global__ __launch_bounds__(256) void head2_kernel(const float* __restrict__ h1,
                                                    const float* __restrict__ w2,
                                                    const float* __restrict__ b2,
                                                    float* __restrict__ logits) {
  int row = blockIdx.x, t = threadIdx.x;
  __shared__ float red[256];
  red[t] = h1[(size_t)row * D_ + t] * w2[t];
  __syncthreads();
  for (int o = 128; o; o >>= 1) { if (t < o) red[t] += red[t + o]; __syncthreads(); }
  if (t == 0) logits[row] = red[0] + b2[0];
}

// ===================== zero fill (float4) =====================
__global__ __launch_bounds__(256) void zero4_kernel(float4* __restrict__ dst) {
  dst[blockIdx.x * 256 + threadIdx.x] = make_float4(0.f, 0.f, 0.f, 0.f);
}

// ===================== 4x4 deconv paint + scatter =====================
__global__ __launch_bounds__(256) void decode_kernel(const float* __restrict__ logits,
                                                     const float* __restrict__ dec_w,
                                                     const float* __restrict__ dec_b,
                                                     const int* __restrict__ idx,
                                                     float* __restrict__ plog,
                                                     float* __restrict__ finalout) {
  int bk = blockIdx.x, b = bk >> 4;
  int p = idx[bk], ph = p >> 6, pw = p & 63;
  int t = threadIdx.x, y = t >> 4, x = t & 15;
  float lg = logits[(bk << 4) + ((y >> 2) << 2) + (x >> 2)];
  float v = lg * dec_w[((y & 3) << 2) + (x & 3)] + dec_b[0];
  plog[(size_t)bk * 256 + t] = v;
  finalout[((size_t)b * HGT + ph * PS_ + y) * WID + pw * PS_ + x] = v;
}

static inline int cdiv(int a, int b) { return (a + b - 1) / b; }

extern "C" void kernel_launch(void* const* d_in, const int* in_sizes, int n_in,
                              void* d_out, int out_size, void* d_ws, size_t ws_size,
                              hipStream_t stream) {
  (void)in_sizes; (void)n_in; (void)out_size; (void)ws_size;
  const float* image  = (const float*)d_in[0];
  const float* labels = (const float*)d_in[1];
  const float* pe_w   = (const float*)d_in[2];
  const float* pe_b   = (const float*)d_in[3];
  const float* pos    = (const float*)d_in[4];
  const float* ipw    = (const float*)d_in[5];
  const float* ipb    = (const float*)d_in[6];
  const float* outw   = (const float*)d_in[7];
  const float* outb   = (const float*)d_in[8];
  const float* ln1s   = (const float*)d_in[9];
  const float* ln1b   = (const float*)d_in[10];
  const float* ff1w   = (const float*)d_in[11];
  const float* ff1b   = (const float*)d_in[12];
  const float* ff2w   = (const float*)d_in[13];
  const float* ff2b   = (const float*)d_in[14];
  const float* ln2s   = (const float*)d_in[15];
  const float* ln2b   = (const float*)d_in[16];
  const float* hlns   = (const float*)d_in[17];
  const float* hlnb   = (const float*)d_in[18];
  const float* hw1    = (const float*)d_in[19];
  const float* hb1    = (const float*)d_in[20];
  const float* hw2    = (const float*)d_in[21];
  const float* hb2    = (const float*)d_in[22];
  const float* decw   = (const float*)d_in[23];
  const float* decb   = (const float*)d_in[24];

  float* out = (float*)d_out;
  char*  ws  = (char*)d_ws;
  float*    scores = (float*)(ws + WS_SCORES);
  int*      idx    = (int*)  (ws + WS_IDX);
  float*    logits = (float*)(ws + WS_LOGITS);
  float*    tokf   = (float*)(ws + WS_TOKF);
  _Float16* tokh   = (_Float16*)(ws + WS_TOKH);
  float*    attnf  = (float*)(ws + WS_ATTNF);
  _Float16* attnh  = (_Float16*)(ws + WS_ATTNH);
  float*    scrf   = (float*)(ws + WS_SCRF);
  _Float16* scrh   = (_Float16*)(ws + WS_SCRH);
  float*    bigf   = (float*)(ws + WS_BIGF);
  _Float16* bigh   = (_Float16*)(ws + WS_BIGH);
  _Float16* wqkv16 = (_Float16*)(ws + WS_W_QKV);
  _Float16* wout16 = (_Float16*)(ws + WS_W_OUT);
  _Float16* wff116 = (_Float16*)(ws + WS_W_FF1);
  _Float16* wff216 = (_Float16*)(ws + WS_W_FF2);
  _Float16* wh116  = (_Float16*)(ws + WS_W_H1);

  // global branch
  coarse_kernel<<<4096, 256, 0, stream>>>(labels, out + OUT_COARSE);
  scores_kernel<<<256, 256, 0, stream>>>(out + OUT_COARSE, scores);
  topk_kernel<<<B_, 256, 0, stream>>>(scores, idx, out + OUT_COORD);
  gather_kernel<<<B_ * KSEL, 256, 0, stream>>>(image, labels, idx,
                                               out + OUT_PATCH, out + OUT_PLAB);

  // weight casts to f16 for WMMA
  cast16_kernel<<<cdiv(NL_ * 3 * D_ * D_, 256), 256, 0, stream>>>(ipw, wqkv16, NL_ * 3 * D_ * D_);
  cast16_kernel<<<cdiv(NL_ * D_ * D_, 256), 256, 0, stream>>>(outw, wout16, NL_ * D_ * D_);
  cast16_kernel<<<cdiv(NL_ * DFF_ * D_, 256), 256, 0, stream>>>(ff1w, wff116, NL_ * DFF_ * D_);
  cast16_kernel<<<cdiv(NL_ * D_ * DFF_, 256), 256, 0, stream>>>(ff2w, wff216, NL_ * D_ * DFF_);
  cast16_kernel<<<cdiv(D_ * D_, 256), 256, 0, stream>>>(hw1, wh116, D_ * D_);

  // local branch: patch embed -> tokens (M = B*S = 4096 rows, D = 256)
  embed_kernel<<<B_ * S_, 256, 0, stream>>>(out + OUT_PATCH, pe_w, pe_b, pos, tokf, tokh);

  const int M = B_ * S_;  // 4096
  for (int l = 0; l < NL_; ++l) {
    gemm_wmma_kernel<<<(M / 32) * (3 * D_ / 64), 32, 0, stream>>>(
        tokh, wqkv16 + (size_t)l * 3 * D_ * D_, ipb + l * 3 * D_, bigf, M, 3 * D_, D_, 0);
    attn_kernel<<<B_ * NH_ * S_, 256, 0, stream>>>(bigf, attnf);
    cast16_kernel<<<cdiv(M * D_, 256), 256, 0, stream>>>(attnf, attnh, M * D_);
    gemm_wmma_kernel<<<(M / 32) * (D_ / 64), 32, 0, stream>>>(
        attnh, wout16 + (size_t)l * D_ * D_, outb + l * D_, scrf, M, D_, D_, 0);
    addln_kernel<<<M, 256, 0, stream>>>(tokf, scrf, ln1s + l * D_, ln1b + l * D_, tokf, tokh);
    gemm_wmma_kernel<<<(M / 32) * (DFF_ / 64), 32, 0, stream>>>(
        tokh, wff116 + (size_t)l * DFF_ * D_, ff1b + l * DFF_, bigf, M, DFF_, D_, 1);
    cast16_kernel<<<cdiv(M * DFF_, 256), 256, 0, stream>>>(bigf, bigh, M * DFF_);
    gemm_wmma_kernel<<<(M / 32) * (D_ / 64), 32, 0, stream>>>(
        bigh, wff216 + (size_t)l * D_ * DFF_, ff2b + l * D_, scrf, M, D_, DFF_, 0);
    addln_kernel<<<M, 256, 0, stream>>>(tokf, scrf, ln2s + l * D_, ln2b + l * D_, tokf, tokh);
  }

  // head
  addln_kernel<<<M, 256, 0, stream>>>(tokf, nullptr, hlns, hlnb, scrf, scrh);
  gemm_wmma_kernel<<<(M / 32) * (D_ / 64), 32, 0, stream>>>(scrh, wh116, hb1, attnf,
                                                            M, D_, D_, 1);
  head2_kernel<<<M, 256, 0, stream>>>(attnf, hw2, hb2, logits);

  // final logit map: zero then scatter painted patches (top-k indices are distinct)
  zero4_kernel<<<16384, 256, 0, stream>>>((float4*)(out + OUT_FINAL));
  decode_kernel<<<B_ * KSEL, 256, 0, stream>>>(logits, decw, decb, idx,
                                               out + OUT_PLOG, out + OUT_FINAL);
}
